// Attention_47708496724821
// MI455X (gfx1250) — matrix operand order repaired and checked by reading, not compiled, not run on hardware
//
#include <hip/hip_runtime.h>
#include <hip/hip_bf16.h>
#include <math.h>

#define HIDDEN 1024
#define BATCH  32
#define SEQ    2048

#define ROWS 128          // encoder rows staged per block (8 waves x 16-row m-tile)
#define LDA  1032         // 1024 + 8 bf16 pad -> 4-bank row skew for ds_load_b128

typedef __attribute__((ext_vector_type(16))) __bf16 v16bf;
typedef __attribute__((ext_vector_type(8)))  __bf16 v8bf;
typedef __attribute__((ext_vector_type(4)))  __bf16 v4bf;
typedef __attribute__((ext_vector_type(8)))  float  v8f;

__device__ __forceinline__ float fast_tanh(float x) {
#if defined(__AMDGCN__) && __has_builtin(__builtin_amdgcn_tanhf)
    return __builtin_amdgcn_tanhf(x);
#elif defined(__AMDGCN__) && __has_builtin(__builtin_amdgcn_tanh_f32)
    return __builtin_amdgcn_tanh_f32(x);
#else
    return tanhf(x);
#endif
}

// ---------------------------------------------------------------------------
// Kernel 1: convert W (f32, row-major [out=1024][in=1024]) to bf16 in workspace
// ---------------------------------------------------------------------------
__global__ void wconv_kernel(const float* __restrict__ W, __bf16* __restrict__ Wb) {
    int i = blockIdx.x * blockDim.x + threadIdx.x;   // one float4 per thread
    float4 f = ((const float4*)W)[i];
    v4bf o;
    o[0] = (__bf16)f.x; o[1] = (__bf16)f.y; o[2] = (__bf16)f.z; o[3] = (__bf16)f.w;
    *(v4bf*)(Wb + (size_t)i * 4) = o;
}

// ---------------------------------------------------------------------------
// Kernel 2: fused  scores[b,s] = v . tanh(enc[b,s,:] @ W^T + dec[b,:])
// Block = 256 threads (8 wave32). Stages 128 encoder rows (full K) as bf16 in
// LDS (258 KB of CDNA5's 320KB WGP LDS). Each wave owns one 16-row m-tile and
// sweeps the 64 n-tiles in PAIRS: one A fragment feeds two independent
// v_wmma_f32_16x16x32_bf16 accumulator chains (halves LDS traffic per WMMA,
// 2x WMMA ILP). Each D tile folds straight into 8 per-lane score
// accumulators -> proj (268 MB) never touches memory.
// ---------------------------------------------------------------------------
__global__ __launch_bounds__(256, 1) void scores_kernel(
    const float*  __restrict__ enc,
    const float*  __restrict__ dec,
    const __bf16* __restrict__ Wb,
    const float*  __restrict__ vvec,
    float*        __restrict__ scores)
{
    extern __shared__ __bf16 As[];   // [ROWS][LDA]

    const int    tid  = threadIdx.x;
    const size_t row0 = (size_t)blockIdx.x * ROWS;        // 128 | 2048 -> one batch per block
    const int    b    = (int)(row0 / SEQ);

    // ---- stage A: 128 rows x 1024 f32 -> bf16 LDS, coalesced float4 loads ----
    const float4* src = (const float4*)(enc + row0 * HIDDEN);
    for (int i = tid; i < ROWS * HIDDEN / 4; i += 256) {
        float4 f = src[i];
        int e = i * 4;
        int r = e >> 10;              // /HIDDEN
        int c = e & (HIDDEN - 1);
        v4bf o;
        o[0] = (__bf16)f.x; o[1] = (__bf16)f.y; o[2] = (__bf16)f.z; o[3] = (__bf16)f.w;
        *(v4bf*)(&As[r * LDA + c]) = o;
    }
    __syncthreads();

    const int wave = tid >> 5;
    const int lane = tid & 31;
    const int m15  = lane & 15;       // A: row within m-tile | B/D: column N
    const int hlf  = lane >> 4;       // half-wave select per ISA layouts

    const __bf16* Arow = &As[(wave * 16 + m15) * LDA];

    float acc[8] = {0.f, 0.f, 0.f, 0.f, 0.f, 0.f, 0.f, 0.f};

    for (int ntp = 0; ntp < HIDDEN / 32; ++ntp) {        // n-tile PAIRS
        const int n0 = ntp * 32 + m15;
        const int n1 = n0 + 16;
        // B[k][n] = W[n][k]: lane n reads 16 contiguous bf16 of W row n
        const __bf16* B0 = Wb + (size_t)n0 * HIDDEN + hlf * 16;
        const __bf16* B1 = Wb + (size_t)n1 * HIDDEN + hlf * 16;

        v8f d0 = {0.f, 0.f, 0.f, 0.f, 0.f, 0.f, 0.f, 0.f};
        v8f d1 = {0.f, 0.f, 0.f, 0.f, 0.f, 0.f, 0.f, 0.f};
        #pragma unroll 4
        for (int k = 0; k < HIDDEN; k += 32) {
            // A 16x32 bf16, contiguous k order used here (same as B): half 0 -> K 0..15, half 1 -> K 16..31
            v8bf alo = *(const v8bf*)(Arow + k + hlf * 16);
            v8bf ahi = *(const v8bf*)(Arow + k + hlf * 16 + 8);
            v16bf a = __builtin_shufflevector(alo, ahi,
                        0, 1, 2, 3, 4, 5, 6, 7, 8, 9, 10, 11, 12, 13, 14, 15);
            v16bf b0 = *(const v16bf*)(B0 + k);
            v16bf b1 = *(const v16bf*)(B1 + k);
            d0 = __builtin_amdgcn_wmma_f32_16x16x32_bf16(
                     false, a, false, b0, (short)0, d0, false, false);
            d1 = __builtin_amdgcn_wmma_f32_16x16x32_bf16(
                     false, a, false, b1, (short)0, d1, false, false);
            if ((k & 96) == 0) {
                // warm next n-tile pair's W lines (L2 -> WGP$); speculative,
                // OOB on the last iteration is silently dropped per ISA
                __builtin_prefetch(B0 + 32 * HIDDEN + k, 0, 1);
                __builtin_prefetch(B1 + 32 * HIDDEN + k, 0, 1);
            }
        }

        // epilogue: fold both D tiles into score accumulators
        // D layout: lane column N, VGPR r -> row M = hlf*8 + r
        const float dv0 = dec[b * HIDDEN + n0];
        const float vv0 = vvec[n0];
        const float dv1 = dec[b * HIDDEN + n1];
        const float vv1 = vvec[n1];
        #pragma unroll
        for (int r = 0; r < 8; ++r)
            acc[r] += vv0 * fast_tanh(d0[r] + dv0)
                    + vv1 * fast_tanh(d1[r] + dv1);
    }

    // reduce over N across the 16 lanes of each half-wave (xor stays in half)
    #pragma unroll
    for (int r = 0; r < 8; ++r) {
        float x = acc[r];
        x += __shfl_xor(x, 1, 32);
        x += __shfl_xor(x, 2, 32);
        x += __shfl_xor(x, 4, 32);
        x += __shfl_xor(x, 8, 32);
        acc[r] = x;
    }
    if (m15 == 0) {
        float* out = scores + row0 + wave * 16 + hlf * 8;
        #pragma unroll
        for (int r = 0; r < 8; ++r) out[r] = acc[r];
    }
}

// ---------------------------------------------------------------------------
// Kernel 3: softmax over S per batch row; also zeroes the context output
// region (d_out is poisoned by the harness; context kernel uses atomics).
// ---------------------------------------------------------------------------
__global__ void softmax_kernel(const float* __restrict__ scores, float* __restrict__ out) {
    const int b   = blockIdx.x;
    const int tid = threadIdx.x;
    __shared__ float red[256];

    const float* srow = scores + (size_t)b * SEQ;
    float lv[8];
    float lmax = -INFINITY;
    #pragma unroll
    for (int i = 0; i < 8; ++i) { lv[i] = srow[tid + i * 256]; lmax = fmaxf(lmax, lv[i]); }

    red[tid] = lmax; __syncthreads();
    for (int s = 128; s > 0; s >>= 1) {
        if (tid < s) red[tid] = fmaxf(red[tid], red[tid + s]);
        __syncthreads();
    }
    const float mx = red[0]; __syncthreads();

    float lsum = 0.f;
    #pragma unroll
    for (int i = 0; i < 8; ++i) { lv[i] = __expf(lv[i] - mx); lsum += lv[i]; }
    red[tid] = lsum; __syncthreads();
    for (int s = 128; s > 0; s >>= 1) {
        if (tid < s) red[tid] += red[tid + s];
        __syncthreads();
    }
    const float inv = 1.0f / red[0];

    float* wrow = out + (size_t)BATCH * HIDDEN + (size_t)b * SEQ;
    #pragma unroll
    for (int i = 0; i < 8; ++i) wrow[tid + i * 256] = lv[i] * inv;

    float* ctx = out + (size_t)b * HIDDEN;
    for (int i = tid; i < HIDDEN; i += 256) ctx[i] = 0.f;
}

// ---------------------------------------------------------------------------
// Kernel 4: context[b,h] = sum_s w[b,s] * enc[b,s,h]  (memory-bound, 268 MB)
// grid = (32 batches, 16 seq-chunks); float4-coalesced; f32 global atomics.
// ---------------------------------------------------------------------------
__global__ void context_kernel(const float* __restrict__ enc,
                               const float* __restrict__ wts,
                               float*       __restrict__ ctx)
{
    const int b  = blockIdx.x;
    const int sc = blockIdx.y;
    const int h  = threadIdx.x * 4;
    const int S0 = sc * (SEQ / 16);

    const float* w = wts + (size_t)b * SEQ + S0;
    const float* e = enc + ((size_t)b * SEQ + S0) * HIDDEN + h;

    float4 a = {0.f, 0.f, 0.f, 0.f};
    for (int s = 0; s < SEQ / 16; ++s) {
        const float  ws = w[s];
        const float4 ev = *(const float4*)(e + (size_t)s * HIDDEN);
        a.x += ws * ev.x; a.y += ws * ev.y; a.z += ws * ev.z; a.w += ws * ev.w;
    }
    float* o = ctx + (size_t)b * HIDDEN + h;
    atomicAdd(o + 0, a.x);
    atomicAdd(o + 1, a.y);
    atomicAdd(o + 2, a.z);
    atomicAdd(o + 3, a.w);
}

// ---------------------------------------------------------------------------
extern "C" void kernel_launch(void* const* d_in, const int* in_sizes, int n_in,
                              void* d_out, int out_size, void* d_ws, size_t ws_size,
                              hipStream_t stream) {
    const float* dec = (const float*)d_in[0];   // [32,1024]
    const float* enc = (const float*)d_in[1];   // [32,2048,1024]
    const float* W   = (const float*)d_in[2];   // [1024,1024]
    const float* v   = (const float*)d_in[3];   // [1,1024]
    float* out = (float*)d_out;                 // context [32,1024] ++ weights [32,2048]

    __bf16* Wb     = (__bf16*)d_ws;                                            // 2 MB
    float*  scores = (float*)((char*)d_ws + (size_t)HIDDEN * HIDDEN * sizeof(__bf16)); // 256 KB

    // 1) W -> bf16
    wconv_kernel<<<(HIDDEN * HIDDEN / 4) / 256, 256, 0, stream>>>(W, Wb);

    // 2) fused WMMA scores
    const size_t lds = (size_t)ROWS * LDA * sizeof(__bf16);   // ~258 KB dynamic LDS
    scores_kernel<<<(BATCH * SEQ) / ROWS, 256, lds, stream>>>(enc, dec, Wb, v, scores);

    // 3) softmax -> weights (into d_out), zero context region
    softmax_kernel<<<BATCH, 256, 0, stream>>>(scores, out);

    // 4) context accumulation
    dim3 g(BATCH, 16);
    context_kernel<<<g, 256, 0, stream>>>(enc, out + (size_t)BATCH * HIDDEN, out);
}
